// ReuseModel_37383395344628
// MI455X (gfx1250) — matrix operand order repaired and checked by reading, not compiled, not run on hardware
//
#include <hip/hip_runtime.h>
#include <hip/hip_bf16.h>

// ---------------------------------------------------------------------------
// Transformer block (LN1+QKV, cosine-gated cache reuse, 16-head attention,
// O-proj, LN2, quickGELU MLP) for MI455X (gfx1250, wave32, WMMA).
// GEMMs + attention use v_wmma_f32_16x16x32_f16 (f16 in, f32 accumulate).
// GEMM A-operand is staged into LDS by the Tensor Data Mover
// (tensor_load_to_lds, TENSORcnt, double-buffered).
// ---------------------------------------------------------------------------

#define B_   16
#define N_   257
#define D_   1024
#define H_   16
#define HD_  64
#define F_   4096
#define M_   (B_ * N_)      // 4112 rows, divisible by 16
#define NP_  272            // padded kv length (multiple of 16)
#define EPS_ 1e-5f
#define SCALE_ 0.125f       // 64^-0.5
#define GINV_  4.0f         // 1 / gating temperature (0.25)

typedef __attribute__((ext_vector_type(16))) _Float16 v16h;
typedef __attribute__((ext_vector_type(8)))  _Float16 v8h;
typedef __attribute__((ext_vector_type(8)))  float    v8f;
typedef __attribute__((ext_vector_type(4)))  unsigned int u32x4;
typedef __attribute__((ext_vector_type(8)))  int      i32x8;
typedef __attribute__((ext_vector_type(4)))  int      i32x4;

// --- WMMA helpers -----------------------------------------------------------

__device__ __forceinline__ v8f wmma_f16(v16h a, v16h b, v8f c) {
  // 8 args: (neg_a, A, neg_b, B, c_mod, C, reuse_a, reuse_b)
  return __builtin_amdgcn_wmma_f32_16x16x32_f16(false, a, false, b,
                                                (short)0, c, false, false);
}

// A-matrix fragment (16x32 f16, MxK). Lane (0..31): m = lane&15, kh = lane>>4.
// Element e -> K = e + kh*16  => one contiguous 16-half chunk
// at (base + kh*16), read as two 8-half loads.
__device__ __forceinline__ v16h load_a_frag(const _Float16* p) {
  v8h lo = *(const v8h*)(p);
  v8h hi = *(const v8h*)(p + 8);
  v16h r;
#pragma unroll
  for (int i = 0; i < 8; ++i) { r[i] = lo[i]; r[i + 8] = hi[i]; }
  return r;
}

// B-matrix fragment (32x16 f16, KxN). Lane column n = lane&15; lanes 0-15 hold
// K=0..15, lanes 16-31 hold K=16..31 => one contiguous 16-half chunk at
// (column_base + kh*16). Caller passes the fully-resolved pointer.
__device__ __forceinline__ v16h load_b_frag(const _Float16* p) {
  return *(const v16h*)(p);
}

// --- Tensor Data Mover: 2-D f16 tile (32 wide x 128 rows) -> LDS ------------
// D# per CDNA5 ISA ch.8: group0 = {count, lds_addr, global_addr[56:0], type=2}
// group1 = {data_size=2B, tensor_dim0=K, tensor_dim1=rows, tile 32x128,
//           tensor_dim0_stride=K}.  Rows past `rows` are zero-filled by TDM.
// This toolchain's builtin is the 6-arg (clang-23 / therock-10.0) form:
// (u32x4 g0, i32x8 g1, i32x4 g2, i32x4 g3, i32x8 extra, i32 cpol).

__device__ __forceinline__ unsigned lds_addr_of(const void* p) {
  // LDS aperture: generic address low 32 bits == LDS byte offset (ISA 10.2)
  return (unsigned)(size_t)p;
}

__device__ __forceinline__ void tdm_load_a_tile(const _Float16* gptr,
                                                unsigned lds_byte_addr,
                                                int K, int rows) {
  unsigned long long ga = (unsigned long long)(size_t)gptr;
  u32x4 g0;
  g0[0] = 1u;                                   // count = 1 valid descriptor
  g0[1] = lds_byte_addr;                        // D#.lds_addr
  g0[2] = (unsigned)(ga & 0xffffffffu);         // global_addr[31:0]
  g0[3] = (unsigned)((ga >> 32) & 0x01ffffffu)  // global_addr[56:32]
        | (2u << 30);                           // type = 2 ("image")
  i32x8 g1;
  g1[0] = (int)(1u << 16);                      // data_size = 1 -> 2 bytes
  g1[1] = (int)(((unsigned)K & 0xffffu) << 16); // tensor_dim0[15:0] @ bit48
  g1[2] = (int)((((unsigned)K >> 16) & 0xffffu)
        | (((unsigned)rows & 0xffffu) << 16));  // dim0 hi | tensor_dim1 lo
  g1[3] = (int)((((unsigned)rows >> 16) & 0xffffu)
        | (32u << 16));                         // dim1 hi | tile_dim0 = 32
  g1[4] = 128;                                  // tile_dim1 = 128, tile_dim2=0
  g1[5] = K;                                    // tensor_dim0_stride[31:0]
  g1[6] = 0;                                    // stride hi | dim1_stride lo
  g1[7] = 0;
  i32x4 z4 = {0, 0, 0, 0};                      // groups 2/3 unused (2-D)
  i32x8 z8 = {0, 0, 0, 0, 0, 0, 0, 0};          // unused trailing group
  __builtin_amdgcn_tensor_load_to_lds(g0, g1, z4, z4, z8, 0);
}

// --- block reduction --------------------------------------------------------

__device__ __forceinline__ float block_reduce_256(float v, float* buf) {
  int t = threadIdx.x;
  buf[t] = v;
  __syncthreads();
#pragma unroll
  for (int s = 128; s > 0; s >>= 1) {
    if (t < s) buf[t] += buf[t + s];
    __syncthreads();
  }
  float r = buf[0];
  __syncthreads();
  return r;
}

// --- weight convert + transpose: wt[n*K + k] = (f16) w[k*Nn + n] ------------

__global__ void wconv_t_kernel(const float* __restrict__ w,
                               _Float16* __restrict__ wt, int K, int Nn) {
  size_t idx = (size_t)blockIdx.x * 256 + threadIdx.x;
  size_t total = (size_t)K * Nn;
  if (idx >= total) return;
  int n = (int)(idx / K);
  int k = (int)(idx % K);
  wt[idx] = (_Float16)w[(size_t)k * Nn + n];
}

// --- LN1 + cosine-sim gate + blended hidden ---------------------------------

__global__ __launch_bounds__(256)
void ln1_gate_kernel(const float* __restrict__ hs, const float* __restrict__ ch,
                     const float* __restrict__ g1, const float* __restrict__ b1,
                     const float* __restrict__ thr,
                     _Float16* __restrict__ hln, float* __restrict__ gate,
                     float* __restrict__ hbl) {
  __shared__ float buf[256];
  int row = blockIdx.x;
  const float* h = hs + (size_t)row * D_;
  const float* c = ch + (size_t)row * D_;
  float sh = 0.f, shh = 0.f, sch = 0.f, scc = 0.f;
  for (int i = threadIdx.x; i < D_; i += 256) {
    float x = h[i], y = c[i];
    sh += x; shh += x * x; sch += x * y; scc += y * y;
  }
  sh  = block_reduce_256(sh, buf);
  shh = block_reduce_256(shh, buf);
  sch = block_reduce_256(sch, buf);
  scc = block_reduce_256(scc, buf);

  float mu = sh * (1.0f / D_);
  float var = shh * (1.0f / D_) - mu * mu;
  float rstd = rsqrtf(var + EPS_);
  float sim = sch / (sqrtf(scc) * sqrtf(shh) + 1e-8f);
  float g = 1.0f / (1.0f + __expf(-(sim - thr[0]) * GINV_));
  if (threadIdx.x == 0) gate[row] = g;

  for (int i = threadIdx.x; i < D_; i += 256) {
    float x = h[i];
    hln[(size_t)row * D_ + i] = (_Float16)((x - mu) * rstd * g1[i] + b1[i]);
    hbl[(size_t)row * D_ + i] = g * c[i] + (1.0f - g) * x;
  }
}

// --- LN2 --------------------------------------------------------------------

__global__ __launch_bounds__(256)
void ln2_kernel(const float* __restrict__ x, const float* __restrict__ g,
                const float* __restrict__ b, _Float16* __restrict__ out) {
  __shared__ float buf[256];
  int row = blockIdx.x;
  const float* p = x + (size_t)row * D_;
  float sh = 0.f, shh = 0.f;
  for (int i = threadIdx.x; i < D_; i += 256) {
    float t = p[i];
    sh += t; shh += t * t;
  }
  sh  = block_reduce_256(sh, buf);
  shh = block_reduce_256(shh, buf);
  float mu = sh * (1.0f / D_);
  float rstd = rsqrtf(shh * (1.0f / D_) - mu * mu + EPS_);
  for (int i = threadIdx.x; i < D_; i += 256)
    out[(size_t)row * D_ + i] = (_Float16)((p[i] - mu) * rstd * g[i] + b[i]);
}

// --- V transpose: vT[b][h][c][n] (NP_-padded, zeros past N) -----------------

__global__ void vtrans_kernel(const _Float16* __restrict__ v,
                              _Float16* __restrict__ vT) {
  size_t idx = (size_t)blockIdx.x * 256 + threadIdx.x;
  size_t total = (size_t)B_ * H_ * HD_ * NP_;
  if (idx >= total) return;
  int n = (int)(idx % NP_);
  int c = (int)((idx / NP_) % HD_);
  int h = (int)((idx / ((size_t)NP_ * HD_)) % H_);
  int b = (int)(idx / ((size_t)NP_ * HD_ * H_));
  _Float16 val = (_Float16)0.0f;
  if (n < N_) val = v[((size_t)(b * N_ + n)) * D_ + h * HD_ + c];
  vT[idx] = val;
}

// --- WMMA GEMM with TDM-staged A and fused epilogues ------------------------
// C[M,Nout] = A[M,K] (f16) x Bt[Nout,K]^T (f16) + bias, then epilogue.
// Block: 256 threads = 8 waves; block tile 128x64; wave tile 16x64.
// A 128x32 tile per K-step is DMA'd into LDS by the TDM (double-buffered);
// wave 0 issues the descriptor, s_wait_tensorcnt + barrier hand it off.

static constexpr int EPI_QKV   = 0;  // f16 out = (g*cached + (1-g)*(acc+b))*alpha
static constexpr int EPI_OPROJ = 1;  // f32 out = resid + acc + b
static constexpr int EPI_GELU  = 2;  // f16 out = quick_gelu(acc + b)
static constexpr int EPI_RES   = 3;  // f32 out = resid + acc + b

template <int EPI>
__global__ __launch_bounds__(256)
void gemm_wmma_kernel(const _Float16* __restrict__ A,
                      const _Float16* __restrict__ Bt,
                      const float* __restrict__ bias,
                      int M, int Nout, int K,
                      const float* __restrict__ gate,
                      const float* __restrict__ aux, float alpha,
                      _Float16* __restrict__ outH, float* __restrict__ outF) {
  __shared__ __align__(16) _Float16 sA[2][128 * 32];   // double-buffered A tile

  int wave = threadIdx.x >> 5;
  int lane = threadIdx.x & 31;
  int mr = lane & 15;
  int kh = lane >> 4;
  int rowBase = blockIdx.y * 128;
  int m0 = rowBase + wave * 16;
  int n0 = blockIdx.x * 64;

  v8f zero = {0.f, 0.f, 0.f, 0.f, 0.f, 0.f, 0.f, 0.f};
  v8f acc[4] = {zero, zero, zero, zero};

  const _Float16* bcol[4];
#pragma unroll
  for (int t = 0; t < 4; ++t)
    bcol[t] = Bt + (size_t)(n0 + t * 16 + mr) * K + kh * 16;

  int remRows = M - rowBase;            // TDM zero-fills rows >= remRows
  const _Float16* atile = A + (size_t)rowBase * K;
  int nsteps = K / 32;

  if (wave == 0)                        // prologue: DMA tile 0 into buffer 0
    tdm_load_a_tile(atile, lds_addr_of(&sA[0][0]), K, remRows);

  for (int s = 0; s < nsteps; ++s) {
    int buf = s & 1;
    if (wave == 0) {
      if (s + 1 < nsteps) {             // DMA tile s+1 into the other buffer
        tdm_load_a_tile(atile + (size_t)(s + 1) * 32,
                        lds_addr_of(&sA[buf ^ 1][0]), K, remRows);
        __builtin_amdgcn_s_wait_tensorcnt(1);   // tile s complete
      } else {
        __builtin_amdgcn_s_wait_tensorcnt(0);
      }
    }
    __syncthreads();                    // tile s visible to all waves

    v16h a = load_a_frag(&sA[buf][(wave * 16 + mr) * 32 + kh * 16]);
#pragma unroll
    for (int t = 0; t < 4; ++t) {
      if (s + 2 < nsteps) __builtin_prefetch(bcol[t] + (s + 2) * 32, 0, 1);
      v16h b = load_b_frag(bcol[t] + s * 32);
      acc[t] = wmma_f16(a, b, acc[t]);
    }
    __syncthreads();                    // done with buf before TDM reuses it
  }

#pragma unroll
  for (int t = 0; t < 4; ++t) {
    int col = n0 + t * 16 + mr;
    float bv = bias[col];
#pragma unroll
    for (int r = 0; r < 8; ++r) {
      int row = m0 + r + kh * 8;
      if (row >= M) continue;
      size_t o = (size_t)row * Nout + col;
      float v = acc[t][r] + bv;
      if constexpr (EPI == EPI_QKV) {
        float g = gate[row];
        outH[o] = (_Float16)((g * aux[o] + (1.0f - g) * v) * alpha);
      } else if constexpr (EPI == EPI_OPROJ) {
        outF[o] = aux[o] + v;
      } else if constexpr (EPI == EPI_GELU) {
        outH[o] = (_Float16)(v / (1.0f + __expf(-1.702f * v)));
      } else {
        outF[o] = aux[o] + v;
      }
    }
  }
}

// --- Flash attention, one wave per (b, h, 16-query tile) --------------------
// Online softmax over 9 kv-chunks of 32. Scores via 4 WMMA/chunk, P staged
// through LDS (f32 -> f16 A-frag), P*V via 4 WMMA/chunk against vT.

__global__ __launch_bounds__(128)
void attn_wmma_kernel(const _Float16* __restrict__ q,   // [B,N,D], pre-scaled
                      const _Float16* __restrict__ k,   // [B,N,D]
                      const _Float16* __restrict__ vT,  // [B,H,HD,NP_]
                      _Float16* __restrict__ attn) {    // [B,N,D]
  int wave = threadIdx.x >> 5;
  int lane = threadIdx.x & 31;
  int task = blockIdx.x * 4 + wave;          // 4352 tasks exactly
  int qt = task % 17;
  int h  = (task / 17) % H_;
  int b  = task / (17 * H_);
  int mr = lane & 15;
  int kh = lane >> 4;

  __shared__ float    s_raw[4][16][33];
  __shared__ __align__(16) _Float16 s_p[4][16][32];
  __shared__ float    s_max[4][16], s_sum[4][16], s_scl[4][16];

  int qrow = qt * 16 + mr;
  if (qrow > N_ - 1) qrow = N_ - 1;
  const _Float16* qbase = q + ((size_t)(b * N_ + qrow)) * D_ + h * HD_;
  v16h qf0 = load_a_frag(qbase + kh * 16);
  v16h qf1 = load_a_frag(qbase + 32 + kh * 16);

  v8f zero = {0.f, 0.f, 0.f, 0.f, 0.f, 0.f, 0.f, 0.f};
  v8f acc[4] = {zero, zero, zero, zero};

  if (lane < 16) { s_max[wave][lane] = -3.0e38f; s_sum[wave][lane] = 0.f; }

  const _Float16* vbase = vT + ((size_t)(b * H_ + h)) * HD_ * NP_;

  for (int kc = 0; kc < 9; ++kc) {
    __syncthreads();
    // ---- scores: S = Q(16x64) x K^T-chunk(64x32), two 16-col halves
#pragma unroll
    for (int half = 0; half < 2; ++half) {
      int kv = kc * 32 + half * 16 + mr;
      int kvc = kv > N_ - 1 ? N_ - 1 : kv;
      const _Float16* kbase = k + ((size_t)(b * N_ + kvc)) * D_ + h * HD_;
      v16h kf0 = load_b_frag(kbase + kh * 16);
      v16h kf1 = load_b_frag(kbase + 32 + kh * 16);
      v8f s = wmma_f16(qf0, kf0, zero);
      s = wmma_f16(qf1, kf1, s);
      bool oob = (kv >= N_);
#pragma unroll
      for (int r = 0; r < 8; ++r)
        s_raw[wave][r + kh * 8][half * 16 + mr] = oob ? -1.0e30f : s[r];
    }
    __syncthreads();
    // ---- online softmax stats + P (f16) for this chunk, lane m owns row m
    if (lane < 16) {
      float om = s_max[wave][lane];
      float m = om;
#pragma unroll
      for (int j = 0; j < 32; ++j) {
        float x = s_raw[wave][lane][j];
        m = x > m ? x : m;
      }
      float scl = __expf(om - m);
      float sum = s_sum[wave][lane] * scl;
#pragma unroll
      for (int j = 0; j < 32; ++j) {
        float p = __expf(s_raw[wave][lane][j] - m);
        sum += p;
        s_p[wave][lane][j] = (_Float16)p;
      }
      s_max[wave][lane] = m;
      s_sum[wave][lane] = sum;
      s_scl[wave][lane] = scl;
    }
    __syncthreads();
    // ---- rescale output accumulators by exp(old_max - new_max)
#pragma unroll
    for (int r = 0; r < 8; ++r) {
      float f = s_scl[wave][r + kh * 8];
#pragma unroll
      for (int t = 0; t < 4; ++t) acc[t][r] *= f;
    }
    // ---- P x V-chunk: P(16x32) A-frag from LDS, V B-frags from vT
    v16h pf = load_a_frag(&s_p[wave][mr][0] + kh * 16);
#pragma unroll
    for (int t = 0; t < 4; ++t) {
      v16h vf = load_b_frag(vbase + (size_t)(t * 16 + mr) * NP_ +
                            kc * 32 + kh * 16);
      acc[t] = wmma_f16(pf, vf, acc[t]);
    }
  }
  __syncthreads();
  // ---- normalize + write
#pragma unroll
  for (int r = 0; r < 8; ++r) {
    int row = qt * 16 + r + kh * 8;
    if (row < N_) {
      float inv = 1.0f / s_sum[wave][r + kh * 8];
#pragma unroll
      for (int t = 0; t < 4; ++t)
        attn[((size_t)(b * N_ + row)) * D_ + h * HD_ + t * 16 + mr] =
            (_Float16)(acc[t][r] * inv);
    }
  }
}

// ---------------------------------------------------------------------------

extern "C" void kernel_launch(void* const* d_in, const int* in_sizes, int n_in,
                              void* d_out, int out_size, void* d_ws,
                              size_t ws_size, hipStream_t stream) {
  (void)in_sizes; (void)n_in; (void)out_size; (void)ws_size;
  const float* hs   = (const float*)d_in[0];
  const float* ch   = (const float*)d_in[1];
  const float* cq   = (const float*)d_in[2];
  const float* ck   = (const float*)d_in[3];
  const float* cv   = (const float*)d_in[4];
  const float* ln1g = (const float*)d_in[5];
  const float* ln1b = (const float*)d_in[6];
  const float* wq   = (const float*)d_in[7];
  const float* bq   = (const float*)d_in[8];
  const float* wk   = (const float*)d_in[9];
  const float* bk   = (const float*)d_in[10];
  const float* wv   = (const float*)d_in[11];
  const float* bv   = (const float*)d_in[12];
  const float* wo   = (const float*)d_in[13];
  const float* bo   = (const float*)d_in[14];
  const float* ln2g = (const float*)d_in[15];
  const float* ln2b = (const float*)d_in[16];
  const float* w1   = (const float*)d_in[17];
  const float* b1   = (const float*)d_in[18];
  const float* w2   = (const float*)d_in[19];
  const float* b2   = (const float*)d_in[20];
  const float* thr  = (const float*)d_in[21];
  float* out = (float*)d_out;

  size_t off = 0;
  auto ws_alloc = [&](size_t bytes) -> void* {
    void* p = (void*)((char*)d_ws + off);
    off = (off + bytes + 255) & ~((size_t)255);
    return p;
  };
  _Float16* HLN  = (_Float16*)ws_alloc((size_t)M_ * D_ * 2);
  float*    GATE = (float*)   ws_alloc((size_t)M_ * 4);
  float*    HBL  = (float*)   ws_alloc((size_t)M_ * D_ * 4);
  _Float16* QF   = (_Float16*)ws_alloc((size_t)M_ * D_ * 2);
  _Float16* KF   = (_Float16*)ws_alloc((size_t)M_ * D_ * 2);
  _Float16* VF   = (_Float16*)ws_alloc((size_t)M_ * D_ * 2);
  _Float16* VT   = (_Float16*)ws_alloc((size_t)B_ * H_ * HD_ * NP_ * 2);
  _Float16* ATT  = (_Float16*)ws_alloc((size_t)M_ * D_ * 2);
  float*    H2   = (float*)   ws_alloc((size_t)M_ * D_ * 4);
  _Float16* LN2O = (_Float16*)ws_alloc((size_t)M_ * D_ * 2);
  _Float16* MID  = (_Float16*)ws_alloc((size_t)M_ * F_ * 2);
  _Float16* WQT  = (_Float16*)ws_alloc((size_t)D_ * D_ * 2);
  _Float16* WKT  = (_Float16*)ws_alloc((size_t)D_ * D_ * 2);
  _Float16* WVT  = (_Float16*)ws_alloc((size_t)D_ * D_ * 2);
  _Float16* WOT  = (_Float16*)ws_alloc((size_t)D_ * D_ * 2);
  _Float16* W1T  = (_Float16*)ws_alloc((size_t)D_ * F_ * 2);
  _Float16* W2T  = (_Float16*)ws_alloc((size_t)F_ * D_ * 2);

  auto launch_w = [&](const float* src, _Float16* dst, int K, int Nn) {
    size_t total = (size_t)K * Nn;
    int blocks = (int)((total + 255) / 256);
    wconv_t_kernel<<<blocks, 256, 0, stream>>>(src, dst, K, Nn);
  };
  launch_w(wq, WQT, D_, D_);
  launch_w(wk, WKT, D_, D_);
  launch_w(wv, WVT, D_, D_);
  launch_w(wo, WOT, D_, D_);
  launch_w(w1, W1T, D_, F_);
  launch_w(w2, W2T, F_, D_);

  ln1_gate_kernel<<<M_, 256, 0, stream>>>(hs, ch, ln1g, ln1b, thr,
                                          HLN, GATE, HBL);

  dim3 gD(D_ / 64, (M_ + 127) / 128);   // (16, 33)
  gemm_wmma_kernel<EPI_QKV><<<gD, 256, 0, stream>>>(
      HLN, WQT, bq, M_, D_, D_, GATE, cq, SCALE_, QF, nullptr);
  gemm_wmma_kernel<EPI_QKV><<<gD, 256, 0, stream>>>(
      HLN, WKT, bk, M_, D_, D_, GATE, ck, 1.0f, KF, nullptr);
  gemm_wmma_kernel<EPI_QKV><<<gD, 256, 0, stream>>>(
      HLN, WVT, bv, M_, D_, D_, GATE, cv, 1.0f, VF, nullptr);

  {
    size_t total = (size_t)B_ * H_ * HD_ * NP_;
    vtrans_kernel<<<(int)((total + 255) / 256), 256, 0, stream>>>(VF, VT);
  }

  // 16*16*17 = 4352 wave-tasks, 4 waves per block -> 1088 blocks
  attn_wmma_kernel<<<1088, 128, 0, stream>>>(QF, KF, VT, ATT);

  gemm_wmma_kernel<EPI_OPROJ><<<gD, 256, 0, stream>>>(
      ATT, WOT, bo, M_, D_, D_, nullptr, HBL, 1.0f, nullptr, H2);

  ln2_kernel<<<M_, 256, 0, stream>>>(H2, ln2g, ln2b, LN2O);

  dim3 gF(F_ / 64, (M_ + 127) / 128);   // (64, 33)
  gemm_wmma_kernel<EPI_GELU><<<gF, 256, 0, stream>>>(
      LN2O, W1T, b1, M_, F_, D_, nullptr, nullptr, 1.0f, MID, nullptr);

  gemm_wmma_kernel<EPI_RES><<<gD, 256, 0, stream>>>(
      MID, W2T, b2, M_, D_, F_, nullptr, H2, 1.0f, nullptr, out);
}
